// ShDictRender_41274635714728
// MI455X (gfx1250) — matrix operand order repaired and checked by reading, not compiled, not run on hardware
//
#include <hip/hip_runtime.h>
#include <hip/hip_bf16.h>
#include <math.h>

#define BATCH     2048
#define NINTRS    256
#define VALID     128
#define NUM_ATOMS 256
#define DATA_DIM  28          // 3*9+1
#define NCOLS     224         // DATA_DIM*8
#define NPTS      (BATCH*VALID)

// packed-word geometry: one uint32 holds bf16 for K=2k (low16) and K=2k+1 (high16)
#define KPAIRS    128         // 256 K / 2
#define WORDS_PER_ARRAY (NCOLS * KPAIRS)   // 28672 words per {hi,lo} array
#define SLICE_WORDS     (NCOLS * 16)       // 3584 words per 32-K slice per array

#if defined(__has_builtin)
#  if __has_builtin(__builtin_amdgcn_global_load_async_to_lds_b128)
#    define HAVE_ASYNC_LDS 1
#  endif
#endif
#ifndef HAVE_ASYNC_LDS
#  define HAVE_ASYNC_LDS 0
#endif

typedef __attribute__((ext_vector_type(16))) __bf16 v16bf;
typedef __attribute__((ext_vector_type(8)))  float  v8f;
typedef __attribute__((ext_vector_type(4)))  int    v4i;

union BfVec { unsigned u[8]; uint4 q[2]; v16bf v; };

// split two f32 into packed bf16-hi word and bf16-lo word (a in low16, b in high16)
__device__ __forceinline__ void split2(float a, float b, unsigned& hi, unsigned& lo) {
    unsigned ua = __float_as_uint(a), ub = __float_as_uint(b);
    unsigned ha = ua & 0xFFFF0000u, hb = ub & 0xFFFF0000u;
    hi = (ha >> 16) | hb;
    float la = a - __uint_as_float(ha);
    float lb = b - __uint_as_float(hb);
    lo = (__float_as_uint(la) >> 16) | (__float_as_uint(lb) & 0xFFFF0000u);
}

// ------- kernel 0: atoms f32 -> WMMA-ready packed bf16 hi / lo word arrays -------
// wsB2[0*WORDS + col*128 + kp] = {hi(2kp) , hi(2kp+1)}
// wsB2[1*WORDS + col*128 + kp] = {lo(2kp) , lo(2kp+1)}
__global__ void prep_atoms(const float* __restrict__ atoms, unsigned* __restrict__ wsB2) {
    int o = blockIdx.x * 256 + threadIdx.x;     // 28672 threads
    int col = o >> 7;                           // 0..223
    int kp  = o & 127;                          // K pair
    float a = atoms[(2 * kp)     * NCOLS + col];
    float b = atoms[(2 * kp + 1) * NCOLS + col];
    unsigned hi, lo;
    split2(a, b, hi, lo);
    wsB2[o] = hi;
    wsB2[WORDS_PER_ARRAY + o] = lo;
}

// ------- kernel 1: fused GEMM (bf16x3) + trilinear + SH contraction --------------
__global__ __launch_bounds__(256, 1)
void gemm_interp(const float* __restrict__ queries,
                 const unsigned* __restrict__ wsB2,
                 const float* __restrict__ intrs_pts,
                 const float* __restrict__ rays_d,
                 float* __restrict__ out4 /* NPTS*4: rgb0,rgb1,rgb2,sigma */) {
    __shared__ unsigned ldsB2[2 * SLICE_WORDS];   // hi slice then lo slice (28672 B)
    __shared__ float    ldsI[8][16][DATA_DIM];    // per-wave interp staging

    const int tid  = threadIdx.x;
    const int lane = tid & 31;
    const int wv   = tid >> 5;
    const int tile = blockIdx.x * 8 + wv;        // 16384 M-tiles of 16 rows
    const int m0   = tile * 16;
    const int row  = m0 + (lane & 15);
    const float* qrow = queries + (size_t)row * 256;
    const int offA = (lane >= 16) ? 16 : 0;      // lane-half K offset (contiguous k = 16h + s)
    const bool hiHalf = lane >= 16;
    const int halfW = hiHalf ? 8 : 0;            // B word offset for lane half

    uint4* ldsQ = (uint4*)ldsB2;
    const uint4* wsQ = (const uint4*)wsB2;

    v8f acc[14];
    v8f zv = {0.f,0.f,0.f,0.f,0.f,0.f,0.f,0.f};
#pragma unroll
    for (int t = 0; t < 14; ++t) acc[t] = zv;

    for (int ks = 0; ks < 8; ++ks) {
        const int kb = ks * 32;
        __syncthreads();
        // stage 32-K slice of B (hi + lo): 1792 uint4 / 256 threads
#pragma unroll
        for (int r = 0; r < 7; ++r) {
            int Q   = tid + r * 256;             // 0..1791
            int sel = Q / 896;                   // 0 = hi, 1 = lo
            int rem = Q - sel * 896;
            int col = rem >> 2;
            int sub = rem & 3;
            const uint4* gsrc = wsQ + sel * (WORDS_PER_ARRAY / 4) + col * 32 + ks * 4 + sub;
#if HAVE_ASYNC_LDS
            __builtin_amdgcn_global_load_async_to_lds_b128((v4i*)gsrc, (v4i*)(ldsQ + Q), 0, 0);
#else
            ldsQ[Q] = *gsrc;
#endif
        }
#if HAVE_ASYNC_LDS
        asm volatile("s_wait_asynccnt 0" ::: "memory");
#endif
        __syncthreads();

        // A tile: rows of queries, split to bf16 hi/lo, contiguous k = 16h + s order (same as B)
        float4 f0 = *(const float4*)(qrow + kb + offA + 0);
        float4 f1 = *(const float4*)(qrow + kb + offA + 4);
        float4 f2 = *(const float4*)(qrow + kb + offA + 8);
        float4 f3 = *(const float4*)(qrow + kb + offA + 12);
        BfVec Ah, Al;
        split2(f0.x, f0.y, Ah.u[0], Al.u[0]);
        split2(f0.z, f0.w, Ah.u[1], Al.u[1]);
        split2(f1.x, f1.y, Ah.u[2], Al.u[2]);
        split2(f1.z, f1.w, Ah.u[3], Al.u[3]);
        split2(f2.x, f2.y, Ah.u[4], Al.u[4]);
        split2(f2.z, f2.w, Ah.u[5], Al.u[5]);
        split2(f3.x, f3.y, Ah.u[6], Al.u[6]);
        split2(f3.z, f3.w, Ah.u[7], Al.u[7]);

#pragma unroll
        for (int t = 0; t < 14; ++t) {
            int col = t * 16 + (lane & 15);
            // WMMA-ready operand words straight from LDS: no unpack VALU
            const uint4* bh = (const uint4*)&ldsB2[col * 16 + halfW];
            const uint4* bl = (const uint4*)&ldsB2[SLICE_WORDS + col * 16 + halfW];
            BfVec Bh, Bl;
            Bh.q[0] = bh[0]; Bh.q[1] = bh[1];
            Bl.q[0] = bl[0]; Bl.q[1] = bl[1];

            acc[t] = __builtin_amdgcn_wmma_f32_16x16x32_bf16(false, Ah.v, false, Bh.v,
                                                             (short)0, acc[t], false, false);
            acc[t] = __builtin_amdgcn_wmma_f32_16x16x32_bf16(false, Ah.v, false, Bl.v,
                                                             (short)0, acc[t], false, false);
            acc[t] = __builtin_amdgcn_wmma_f32_16x16x32_bf16(false, Al.v, false, Bh.v,
                                                             (short)0, acc[t], false, false);
        }
    }

    // ---- epilogue: trilinear weights (per-lane c fixed, rows via broadcast) ----
    float fxv = 0.f, fyv = 0.f, fzv = 0.f;
    if (lane < 16) {
        const float* pp = intrs_pts + (size_t)(m0 + lane) * 3;
        float px = pp[0] * 128.f + 1e-5f;
        float py = pp[1] * 128.f + 1e-5f;
        float pz = pp[2] * 128.f + 1e-5f;
        fxv = px - floorf(px); fyv = py - floorf(py); fzv = pz - floorf(pz);
    }
    const int c = lane & 7;
    float w8[16];
#pragma unroll
    for (int i = 0; i < 16; ++i) {
        float xi = __shfl(fxv, i, 32);
        float yi = __shfl(fyv, i, 32);
        float zi = __shfl(fzv, i, 32);
        float gx = (c & 4) ? xi : 1.f - xi;
        float gy = (c & 2) ? yi : 1.f - yi;
        float gz = (c & 1) ? zi : 1.f - zi;
        w8[i] = gx * gy * gz;
    }

#pragma unroll
    for (int t = 0; t < 14; ++t) {
#pragma unroll
        for (int r = 0; r < 8; ++r) {
            float wv_ = hiHalf ? w8[r + 8] : w8[r];
            float p = acc[t][r] * wv_;
            p += __shfl_xor(p, 1, 32);           // reduce over 8-lane c-group
            p += __shfl_xor(p, 2, 32);
            p += __shfl_xor(p, 4, 32);
            if ((lane & 7) == 0) {
                int rowL = r + (hiHalf ? 8 : 0);
                int d = t * 2 + ((lane >> 3) & 1);
                ldsI[wv][rowL][d] = p;
            }
        }
    }
    __syncthreads();

    // SH basis: all 16 rows in a tile share one ray (16 | 128)
    const int ray = m0 >> 7;
    const float* rdp = rays_d + ray * 3;
    float dx = rdp[0], dy = rdp[1], dz = rdp[2];
    float inv = rsqrtf(dx * dx + dy * dy + dz * dz);
    dx *= inv; dy *= inv; dz *= inv;
    float sh[9];
    sh[0] = 0.28209479177387814f;
    sh[1] = -0.4886025119029199f * dy;
    sh[2] =  0.4886025119029199f * dz;
    sh[3] = -0.4886025119029199f * dx;
    sh[4] =  1.0925484305920792f * dx * dy;
    sh[5] = -1.0925484305920792f * dy * dz;
    sh[6] =  0.31539156525252005f * (2.f * dz * dz - dx * dx - dy * dy);
    sh[7] = -1.0925484305920792f * dx * dz;
    sh[8] =  0.5462742152960396f * (dx * dx - dy * dy);

#pragma unroll
    for (int s = 0; s < 2; ++s) {
        int oi = lane + s * 32;                  // 64 outputs: 16 rows x (rgb0..2,sigma)
        int rowL = oi >> 2;
        int o = oi & 3;
        float val;
        if (o == 3) {
            val = ldsI[wv][rowL][27];
        } else {
            val = 0.f;
#pragma unroll
            for (int k = 0; k < 9; ++k) val += sh[k] * ldsI[wv][rowL][o * 9 + k];
        }
        out4[(size_t)(m0 + rowL) * 4 + o] = val;
    }
}

// ------- kernel 2: per-ray volume rendering, wave-scan cumprod -------------------
__global__ __launch_bounds__(256)
void render(const float4* __restrict__ pt4,
            const float* __restrict__ intersections,
            const float* __restrict__ rays_d,
            float* __restrict__ rgb_out,
            float* __restrict__ alpha_out,
            float* __restrict__ depth_out) {
    const int tid = threadIdx.x, lane = tid & 31;
    const int ray = blockIdx.x * 8 + (tid >> 5);
    const float* it = intersections + (size_t)ray * (NINTRS + 1);
    float dx = rays_d[ray * 3 + 0], dy = rays_d[ray * 3 + 1], dz = rays_d[ray * 3 + 2];
    float nrm = sqrtf(dx * dx + dy * dy + dz * dz);

    float carry = 1.f, r0 = 0.f, r1 = 0.f, r2 = 0.f, wa = 0.f, da = 0.f;
#pragma unroll
    for (int chunk = 0; chunk < 4; ++chunk) {
        int i = chunk * 32 + lane;
        float4 p = pt4[(size_t)ray * VALID + i];
        float t0 = it[i], t1 = it[i + 1];
        float dist = (t1 - t0) * nrm;
        float sig = fmaxf(p.w, 0.f);
        float alpha = 1.f - expf(-sig * dist);
        alpha_out[(size_t)ray * NINTRS + i] = alpha;
        float incl = 1.f - alpha + 1e-10f;
#pragma unroll
        for (int off = 1; off < 32; off <<= 1) {   // inclusive scan product
            float v = __shfl_up(incl, off, 32);
            if (lane >= off) incl *= v;
        }
        float excl = __shfl_up(incl, 1, 32);
        if (lane == 0) excl = 1.f;
        float trans = carry * excl;
        float ab = alpha * trans;
        wa += ab;
        da += ab * 0.5f * (t0 + t1);
        r0 += ab / (1.f + expf(-p.x));
        r1 += ab / (1.f + expf(-p.y));
        r2 += ab / (1.f + expf(-p.z));
        carry *= __shfl(incl, 31, 32);
    }
#pragma unroll
    for (int chunk = 4; chunk < 8; ++chunk)        // masked region: alpha == 0
        alpha_out[(size_t)ray * NINTRS + chunk * 32 + lane] = 0.f;
#pragma unroll
    for (int off = 16; off > 0; off >>= 1) {
        r0 += __shfl_xor(r0, off, 32);
        r1 += __shfl_xor(r1, off, 32);
        r2 += __shfl_xor(r2, off, 32);
        wa += __shfl_xor(wa, off, 32);
        da += __shfl_xor(da, off, 32);
    }
    if (lane == 0) {
        float bg = 1.f - wa;                       // WHITE_BKGD
        rgb_out[ray * 3 + 0] = r0 + bg;
        rgb_out[ray * 3 + 1] = r1 + bg;
        rgb_out[ray * 3 + 2] = r2 + bg;
        depth_out[ray] = da;
    }
}

extern "C" void kernel_launch(void* const* d_in, const int* in_sizes, int n_in,
                              void* d_out, int out_size, void* d_ws, size_t ws_size,
                              hipStream_t stream) {
    (void)in_sizes; (void)n_in; (void)out_size; (void)ws_size;
    const float* queries       = (const float*)d_in[0];
    const float* atoms         = (const float*)d_in[1];
    const float* intrs_pts     = (const float*)d_in[2];
    const float* intersections = (const float*)d_in[3];
    const float* rays_d        = (const float*)d_in[4];
    // d_in[5] flat_idx is b*NINTRS+i with i<VALID: gather mapping n = ray*VALID+i is used directly.

    unsigned* wsB2 = (unsigned*)d_ws;                                   // 2*28672 words
    float*    out4 = (float*)((char*)d_ws + (size_t)2 * WORDS_PER_ARRAY * 4);
    float*    out  = (float*)d_out;
    float*    rgb   = out;                          // 2048*3
    float*    alpha = out + BATCH * 3;              // 2048*256
    float*    depth = out + BATCH * 3 + BATCH * NINTRS;

    prep_atoms<<<2 * WORDS_PER_ARRAY / 2 / 256, 256, 0, stream>>>(atoms, wsB2);
    gemm_interp<<<NPTS / 16 / 8, 256, 0, stream>>>(queries, wsB2, intrs_pts, rays_d, out4);
    render<<<BATCH / 8, 256, 0, stream>>>((const float4*)out4, intersections, rays_d,
                                          rgb, alpha, depth);
}
